// LoRAColumnParallelLinear_82660940578996
// MI455X (gfx1250) — compile-verified, matrix operand order checked by reading, not run on hardware
//
#include <hip/hip_runtime.h>
#include <stdint.h>

typedef __attribute__((ext_vector_type(16))) __bf16 v16bf;
typedef __attribute__((ext_vector_type(8)))  float  v8f;
typedef __attribute__((ext_vector_type(4)))  float  f32x4;
typedef __attribute__((ext_vector_type(4)))  int    v4i;

#define M_DIM 4096
#define N_DIM 16384
#define K_DIM 4096
#define R_DIM 16
#define LORA_SCALE 2.0f

#define BM 128
#define BN 128
#define BK 32
#define LDSS 40   // padded LDS row stride (elements): 80B rows, conflict-free b128 frag reads

// CDNA5 async global->LDS path (ASYNCcnt), with full fallback if the builtins
// are not declared by this toolchain.
#if defined(__has_builtin)
#if __has_builtin(__builtin_amdgcn_global_load_async_to_lds_b128) && \
    __has_builtin(__builtin_amdgcn_s_wait_asynccnt)
#define USE_ASYNC_LDS 1
#endif
#endif

#define AS1 __attribute__((address_space(1)))
#define AS3 __attribute__((address_space(3)))

#ifdef USE_ASYNC_LDS
static __device__ __forceinline__ void async_b128(const unsigned short* g,
                                                  unsigned short* l) {
  // param0: int4 in AS1 (global). param1: LDS dest; generic LDS ptr low 32 bits
  // = LDS offset (ISA flat->LDS truncation rule).
  __builtin_amdgcn_global_load_async_to_lds_b128(
      (AS1 v4i*)(uintptr_t)g,
      (AS3 v4i*)(uint32_t)(uintptr_t)l,
      0, 0);
}
#endif

static __device__ __forceinline__ void wait_async_all() {
#ifdef USE_ASYNC_LDS
  __builtin_amdgcn_s_wait_asynccnt(0);
#endif
}

// fp32 -> bf16 round-to-nearest-even
static __device__ __forceinline__ unsigned short f2bf(float f) {
  unsigned int u = __float_as_uint(f);
  unsigned int r = u + 0x7FFFu + ((u >> 16) & 1u);
  return (unsigned short)(r >> 16);
}

// ---------------------------------------------------------------- kernel 1
__global__ void cvt_input_bf16(const float* __restrict__ in,
                               unsigned short* __restrict__ out, int n) {
  int i = (blockIdx.x * blockDim.x + threadIdx.x) * 4;
  if (i >= n) return;
  // read-once fp32: nontemporal load keeps L2 free for the bf16 working set
  f32x4 f = __builtin_nontemporal_load((const f32x4*)(in + i));
  ushort4 o;
  o.x = f2bf(f.x); o.y = f2bf(f.y); o.z = f2bf(f.z); o.w = f2bf(f.w);
  *reinterpret_cast<ushort4*>(out + i) = o;   // RT store: keep bf16 A in L2
}

// ---------------------------------------------------------------- kernel 2
// W'[o,d] = bf16( W[o,d] + 2.0 * sum_r lora_B[o,r] * lora_A[r,d] )
__global__ void fold_weight_bf16(const float* __restrict__ W,
                                 const float* __restrict__ lA,   // [R, K]
                                 const float* __restrict__ lB,   // [N, R]
                                 unsigned short* __restrict__ out) {
  __shared__ float At[R_DIM][64];
  __shared__ float Bt[64][R_DIM];
  const int o0 = blockIdx.x * 64;
  const int d0 = blockIdx.y * 64;
  const int t  = threadIdx.x;

  for (int idx = t; idx < 1024; idx += 256) {
    At[idx >> 6][idx & 63] = lA[(idx >> 6) * K_DIM + d0 + (idx & 63)];
    Bt[idx >> 4][idx & 15] = lB[(size_t)(o0 + (idx >> 4)) * R_DIM + (idx & 15)];
  }
  __syncthreads();

  const int oL  = t & 63;
  const int dL0 = (t >> 6) << 4;
  float b[R_DIM];
#pragma unroll
  for (int r = 0; r < R_DIM; ++r) b[r] = Bt[oL][r];

  const size_t rowBase = (size_t)(o0 + oL) * K_DIM + d0;
#pragma unroll
  for (int j = 0; j < 16; ++j) {
    const int dL = dL0 + j;
    float acc = 0.f;
#pragma unroll
    for (int r = 0; r < R_DIM; ++r) acc += b[r] * At[r][dL];
    float w = __builtin_nontemporal_load(&W[rowBase + dL]) + LORA_SCALE * acc;
    out[rowBase + dL] = f2bf(w);              // RT store: keep bf16 W' in L2
  }
}

// ---------------------------------------------------------------- kernel 3
// C[M,N] (fp32) = A[M,K] (bf16) * B[N,K]^T (bf16), WMMA 16x16x32 bf16.
// Double-buffered LDS, async global->LDS tile feed, K-loop unrolled x2 so the
// ping-pong buffer index is a compile-time constant (no acc register shuffling).
__global__ void __launch_bounds__(256)
lora_gemm_bf16(const unsigned short* __restrict__ A,
               const unsigned short* __restrict__ B,
               float* __restrict__ C) {
  __shared__ unsigned short As[2][BM * LDSS];
  __shared__ unsigned short Bs[2][BN * LDSS];

  const int t     = threadIdx.x;
  const int lane  = t & 31;
  const int wave  = t >> 5;       // 0..7
  const int waveM = wave >> 2;    // 0..1  -> 64 rows each
  const int waveN = wave & 3;     // 0..3  -> 32 cols each
  const int m0    = blockIdx.x * BM;
  const int n0    = blockIdx.y * BN;

  // global->LDS: 2 threads per row, 16 bf16 (32B) each, via 2x b128
  const int ldRow  = t >> 1;          // 0..127
  const int ldCol  = (t & 1) << 4;    // 0 or 16
  const int ldsOff = ldRow * LDSS + ldCol;

  const unsigned short* aSrc = A + (size_t)(m0 + ldRow) * K_DIM + ldCol;
  const unsigned short* bSrc = B + (size_t)(n0 + ldRow) * K_DIM + ldCol;

  const int laneRow  = lane & 15;
  const int laneHalf = lane >> 4;      // 0/1
  const int kOff     = laneHalf << 3;  // frag K chunks: {kOff..+7, kOff+16..+23}

  v8f acc[4][2];
#pragma unroll
  for (int mi = 0; mi < 4; ++mi)
#pragma unroll
    for (int ni = 0; ni < 2; ++ni) {
      v8f z = {};
      acc[mi][ni] = z;
    }

  union Frag { uint4 q[2]; v16bf v; };

  auto issue_tile = [&](int kG, int buf) __attribute__((always_inline)) {
#ifdef USE_ASYNC_LDS
    async_b128(aSrc + kG,     &As[buf][ldsOff]);
    async_b128(aSrc + kG + 8, &As[buf][ldsOff + 8]);
    async_b128(bSrc + kG,     &Bs[buf][ldsOff]);
    async_b128(bSrc + kG + 8, &Bs[buf][ldsOff + 8]);
#else
    uint4 a0 = *reinterpret_cast<const uint4*>(aSrc + kG);
    uint4 a1 = *reinterpret_cast<const uint4*>(aSrc + kG + 8);
    uint4 b0 = *reinterpret_cast<const uint4*>(bSrc + kG);
    uint4 b1 = *reinterpret_cast<const uint4*>(bSrc + kG + 8);
    *reinterpret_cast<uint4*>(&As[buf][ldsOff])     = a0;
    *reinterpret_cast<uint4*>(&As[buf][ldsOff + 8]) = a1;
    *reinterpret_cast<uint4*>(&Bs[buf][ldsOff])     = b0;
    *reinterpret_cast<uint4*>(&Bs[buf][ldsOff + 8]) = b1;
#endif
  };

  auto compute_tile = [&](int buf) __attribute__((always_inline)) {
    Frag aF[4], bF[2];
#pragma unroll
    for (int mi = 0; mi < 4; ++mi) {
      const unsigned short* p =
          &As[buf][(waveM * 64 + mi * 16 + laneRow) * LDSS + kOff];
      aF[mi].q[0] = *reinterpret_cast<const uint4*>(p);
      aF[mi].q[1] = *reinterpret_cast<const uint4*>(p + 16);
    }
#pragma unroll
    for (int ni = 0; ni < 2; ++ni) {
      const unsigned short* p =
          &Bs[buf][(waveN * 32 + ni * 16 + laneRow) * LDSS + kOff];
      bF[ni].q[0] = *reinterpret_cast<const uint4*>(p);
      bF[ni].q[1] = *reinterpret_cast<const uint4*>(p + 16);
    }
#pragma unroll
    for (int mi = 0; mi < 4; ++mi)
#pragma unroll
      for (int ni = 0; ni < 2; ++ni)
        acc[mi][ni] = __builtin_amdgcn_wmma_f32_16x16x32_bf16(
            false, aF[mi].v, false, bF[ni].v, (short)0, acc[mi][ni], false, false);
  };

  constexpr int NK = K_DIM / BK;   // 128, even
  issue_tile(0, 0);

#pragma unroll 1
  for (int kt = 0; kt < NK; kt += 2) {
    // ---- even tile (buffer 0) ----
    wait_async_all();              // my async loads for tile kt landed in LDS
    __syncthreads();               // everyone's landed; buf1 fully consumed
    issue_tile((kt + 1) * BK, 1);  // kt+1 <= NK-1 always (NK even)
    compute_tile(0);

    // ---- odd tile (buffer 1) ----
    wait_async_all();
    __syncthreads();
    if (kt + 2 < NK) issue_tile((kt + 2) * BK, 0);
    compute_tile(1);
  }

  // C layout: VGPR v, lanes 0-15 -> M=v, N=lane; lanes 16-31 -> M=v+8, N=lane-16
  // NT stores: C (256MB) is write-once; keep it out of L2 so A/B stay resident.
  const int nCol  = n0 + waveN * 32 + laneRow;
  const int mBase = m0 + waveM * 64 + (laneHalf << 3);
#pragma unroll
  for (int mi = 0; mi < 4; ++mi)
#pragma unroll
    for (int ni = 0; ni < 2; ++ni)
#pragma unroll
      for (int v = 0; v < 8; ++v)
        __builtin_nontemporal_store(
            acc[mi][ni][v],
            &C[(size_t)(mBase + mi * 16 + v) * N_DIM + (nCol + ni * 16)]);
}

// ---------------------------------------------------------------- kernel 4
__global__ void copy_bias(const float* __restrict__ bias, float* __restrict__ out) {
  int i = blockIdx.x * blockDim.x + threadIdx.x;
  if (i < N_DIM) out[i] = bias[i];
}

// ----------------------------------------------------------------
extern "C" void kernel_launch(void* const* d_in, const int* in_sizes, int n_in,
                              void* d_out, int out_size, void* d_ws, size_t ws_size,
                              hipStream_t stream) {
  const float* input  = (const float*)d_in[0];   // [2,2048,4096]
  const float* weight = (const float*)d_in[1];   // [16384,4096]
  const float* bias   = (const float*)d_in[2];   // [16384]
  const float* loraA  = (const float*)d_in[3];   // [16,4096]
  const float* loraB  = (const float*)d_in[4];   // [16384,16]
  float* out = (float*)d_out;

  // workspace layout: [bf16 A : 32MB][bf16 W' : 128MB]  (fits 192MB L2 together)
  unsigned short* aBf = (unsigned short*)d_ws;
  unsigned short* wBf = aBf + (size_t)M_DIM * K_DIM;

  cvt_input_bf16<<<(M_DIM * K_DIM / 4 + 255) / 256, 256, 0, stream>>>(
      input, aBf, M_DIM * K_DIM);
  fold_weight_bf16<<<dim3(N_DIM / 64, K_DIM / 64), 256, 0, stream>>>(
      weight, loraA, loraB, wBf);
  // x = M blocks (fast), y = N blocks: weight tile rows get L2 reuse across M sweep
  lora_gemm_bf16<<<dim3(M_DIM / BM, N_DIM / BN), 256, 0, stream>>>(aBf, wBf, out);
  copy_bias<<<(N_DIM + 255) / 256, 256, 0, stream>>>(bias, out + (size_t)M_DIM * N_DIM);
}
